// LocEncoder_53008486367321
// MI455X (gfx1250) — matrix operand order repaired and checked by reading, not compiled, not run on hardware
//
#include <hip/hip_runtime.h>
#include <hip/hip_bf16.h>

typedef __attribute__((ext_vector_type(16))) _Float16 v16h;
typedef __attribute__((ext_vector_type(8)))  float    v8f;

#define D_IN   13
#define D_POS  3
#define HIDDEN 64
#define D_OUTC 64
#define WAVES_PER_BLOCK 8
#define TILES_PER_WAVE  8     // amortize weight-fragment gather over 8x16 = 128 edges
#define LDS_STRIDE 72         // 64 halves + 8 pad halves -> breaks bank alignment

__global__ __launch_bounds__(256) void locenc_zero_kernel(unsigned int* out, int n) {
    int i = blockIdx.x * blockDim.x + threadIdx.x;
    if (i < n) out[i] = 0u;   // 0u == bits of +0.0f
}

__global__ __launch_bounds__(256) void locenc_mlp_max_kernel(
    const float* __restrict__ x, const float* __restrict__ pos,
    const float* __restrict__ W1, const float* __restrict__ b1,
    const float* __restrict__ W2, const float* __restrict__ b2,
    const int* __restrict__ eidx, unsigned int* __restrict__ out, int E)
{
    __shared__ _Float16 hbuf[WAVES_PER_BLOCK][16][LDS_STRIDE];

    const int lane = threadIdx.x & 31;
    const int wave = threadIdx.x >> 5;
    const int hi   = lane >> 4;     // 0: lanes 0-15, 1: lanes 16-31
    const int lm   = lane & 15;

    // ============ hoisted, tile-invariant state (loaded once per wave) ============
    // Weight fragments; B layout: lanes0-15 -> K 0..7,16..23, lanes16-31 -> K 8..15,24..31, N = lane&15
    v16h b1f[4];                         // W1: 16x64, K rows 16..31 are zero pad
    #pragma unroll
    for (int n = 0; n < 4; ++n) {
        const int col = n * 16 + lm;
        #pragma unroll
        for (int j = 0; j < 16; ++j) {
            const int k = (j < 8) ? hi * 8 + j : 16 + hi * 8 + (j - 8);
            b1f[n][j] = (k < D_IN + D_POS) ? (_Float16)W1[k * HIDDEN + col] : (_Float16)0.0f;
        }
    }
    v16h b2f[2][4];                      // W2: 64x64, two K-chunks of 32
    #pragma unroll
    for (int kc = 0; kc < 2; ++kc) {
        #pragma unroll
        for (int n = 0; n < 4; ++n) {
            const int col = n * 16 + lm;
            #pragma unroll
            for (int j = 0; j < 16; ++j) {
                const int k = kc * 32 + ((j < 8) ? hi * 8 + j : 16 + hi * 8 + (j - 8));
                b2f[kc][n][j] = (_Float16)W2[k * D_OUTC + col];
            }
        }
    }
    // Bias broadcast values (C layout: every VGPR of a column tile holds same bias)
    float bias1[4], bias2[4];
    #pragma unroll
    for (int n = 0; n < 4; ++n) { bias1[n] = b1[n * 16 + lm]; bias2[n] = b2[n * 16 + lm]; }

    _Float16 (*hrow)[LDS_STRIDE] = hbuf[wave];

    const long long numTiles = ((long long)E + 15) / 16;
    const long long t0 = (long long)(blockIdx.x * WAVES_PER_BLOCK + wave) * TILES_PER_WAVE;

    // ============ per-tile loop: 16 edges per iteration ============
    #pragma unroll 1
    for (int t = 0; t < TILES_PER_WAVE; ++t) {
        long long tile = t0 + t;
        if (tile >= numTiles) tile = numTiles - 1;      // duplicate work, idempotent under max
        long long tileBase = tile * 16;
        if (tileBase > (long long)E - 16) tileBase = (E >= 16) ? (long long)E - 16 : 0;

        // prefetch next tile's edge-index cacheline (global_prefetch_b8)
        long long pfe = tileBase + 16;
        if (pfe > (long long)E - 1) pfe = (long long)E - 1;
        __builtin_prefetch(eidx + pfe, 0, 0);
        __builtin_prefetch(eidx + (long long)E + pfe, 0, 0);

        long long e = tileBase + lm; if (e > (long long)E - 1) e = (long long)E - 1;
        const int src = eidx[e];
        const int dst = eidx[(long long)E + e];

        // Gather message features; A layout (16-bit 16x32):
        //  lanes 0-15 : row M=lm, halves 0..7 -> K 0..7,  halves 8..15 -> K 16..23 (pad)
        //  lanes 16-31: row M=lm, halves 0..7 -> K 8..15, halves 8..15 -> K 24..31 (pad)
        v16h a1;
        #pragma unroll
        for (int j = 0; j < 16; ++j) a1[j] = (_Float16)0.0f;
        if (hi == 0) {
            #pragma unroll
            for (int j = 0; j < 8; ++j)                 // features 0..7 = x[src][0..7]
                a1[j] = (_Float16)x[(long long)src * D_IN + j];
        } else {
            #pragma unroll
            for (int j = 0; j < 5; ++j)                 // features 8..12 = x[src][8..12]
                a1[j] = (_Float16)x[(long long)src * D_IN + 8 + j];
            #pragma unroll
            for (int c = 0; c < 3; ++c)                 // features 13..15 = pos[src]-pos[dst]
                a1[5 + c] = (_Float16)(pos[(long long)src * D_POS + c] -
                                       pos[(long long)dst * D_POS + c]);
        }

        // ---- Layer 1: h = relu(A1 @ W1 + b1), stage as f16 tile in LDS ----
        #pragma unroll
        for (int n = 0; n < 4; ++n) {
            v8f c;
            #pragma unroll
            for (int r = 0; r < 8; ++r) c[r] = bias1[n];
            c = __builtin_amdgcn_wmma_f32_16x16x32_f16(false, a1, false, b1f[n],
                                                       (short)0, c, false, false);
            #pragma unroll
            for (int r = 0; r < 8; ++r) {               // C layout: row = r + 8*hi
                float v = c[r] > 0.0f ? c[r] : 0.0f;
                hrow[r + hi * 8][n * 16 + lm] = (_Float16)v;
            }
        }
        // Wave-local LDS RAW fence: DS pipe is in-order per wave; this orders the
        // stores above against the reads below without a block-wide barrier.
        asm volatile("s_wait_dscnt 0" ::: "memory");

        // ---- Re-load activations in A layout for layer 2 (K = 64 -> 2 chunks) ----
        v16h a2[2];
        {
            const _Float16* row = hrow[lm];             // row M = lm
            #pragma unroll
            for (int kc = 0; kc < 2; ++kc) {
                const int base = kc * 32 + hi * 8;      // K' 0..7 (lo) / 8..15 (hi)
                #pragma unroll
                for (int j = 0; j < 8; ++j) {
                    a2[kc][j]     = row[base + j];      // halves 0..7  -> K' base..base+7
                    a2[kc][8 + j] = row[base + 16 + j]; // halves 8..15 -> K' base+16..+23
                }
            }
        }

        // ---- Layer 2 + clamp(>=0) + bitwise-uint atomic max scatter ----
        #pragma unroll
        for (int n = 0; n < 4; ++n) {
            v8f c;
            #pragma unroll
            for (int r = 0; r < 8; ++r) c[r] = bias2[n];
            c = __builtin_amdgcn_wmma_f32_16x16x32_f16(false, a2[0], false, b2f[0][n],
                                                       (short)0, c, false, false);
            c = __builtin_amdgcn_wmma_f32_16x16x32_f16(false, a2[1], false, b2f[1][n],
                                                       (short)0, c, false, false);
            #pragma unroll
            for (int r = 0; r < 8; ++r) {
                const int m  = r + hi * 8;              // edge row within tile
                const int dm = __shfl(dst, m, 32);      // lane m holds dst for row m
                float v = c[r] > 0.0f ? c[r] : 0.0f;    // fold final relu into the max
                atomicMax(out + (long long)dm * D_OUTC + n * 16 + lm, __float_as_uint(v));
            }
        }
        // Order this iteration's LDS reads before next iteration's stores (WAR).
        asm volatile("" ::: "memory");
    }
}

extern "C" void kernel_launch(void* const* d_in, const int* in_sizes, int n_in,
                              void* d_out, int out_size, void* d_ws, size_t ws_size,
                              hipStream_t stream) {
    const float* x   = (const float*)d_in[0];
    const float* pos = (const float*)d_in[1];
    const float* W1  = (const float*)d_in[2];
    const float* b1  = (const float*)d_in[3];
    const float* W2  = (const float*)d_in[4];
    const float* b2  = (const float*)d_in[5];
    const int*   eix = (const int*)d_in[6];
    const int E = in_sizes[6] / 2;

    // Zero-init output: bit pattern 0 == +0.0f, the identity for the
    // relu-folded max aggregation (empty segments -> 0, matching reference).
    locenc_zero_kernel<<<(out_size + 255) / 256, 256, 0, stream>>>(
        (unsigned int*)d_out, out_size);

    const long long numTiles = ((long long)E + 15) / 16;
    const long long tilesPerBlock = (long long)WAVES_PER_BLOCK * TILES_PER_WAVE;
    const int blocks = (int)((numTiles + tilesPerBlock - 1) / tilesPerBlock);
    locenc_mlp_max_kernel<<<blocks, 256, 0, stream>>>(
        x, pos, W1, b1, W2, b2, eix, (unsigned int*)d_out, E);
}
